// MultiHeadAttention_25460566131169
// MI455X (gfx1250) — matrix operand order repaired and checked by reading, not compiled, not run on hardware
//
#include <hip/hip_runtime.h>
#include <hip/hip_bf16.h>

// MI455X / gfx1250 multi-head attention forward.
// Compute-bound (~100 GFLOP vs ~50MB data) -> WMMA f16 path with f32 accum.
// All matrix math via v_wmma_f32_16x16x32_f16 (wave32, 16x16x32 MACs/instr).
// Weights pre-converted to f16 once; GEMM waves own 32x64 tiles (8 WMMAs per
// 6 fragment loads). Attention K-tiles are moved with double-buffered
// GLOBAL_LOAD_ASYNC_TO_LDS_B128 (ASYNCcnt pipelining). Projection layout/scale
// choices are template parameters -> branch-free epilogues.

typedef __attribute__((ext_vector_type(16))) _Float16 v16h;
typedef __attribute__((ext_vector_type(8)))  _Float16 v8h;
typedef __attribute__((ext_vector_type(8)))  float    v8f;
typedef __attribute__((ext_vector_type(4)))  int      v4i;

#define BATCH   2
#define SEQ     2048
#define DEMB    1024
#define NHEAD   16
#define DHEAD   64
#define MTOT    (BATCH * SEQ)       // 4096 rows in the flattened GEMMs
#define SCALE_INV 0.125f            // 1/sqrt(d_head)

#if __has_builtin(__builtin_amdgcn_global_load_async_to_lds_b128)
#define HAVE_ASYNC_LDS 1
#else
#define HAVE_ASYNC_LDS 0
#endif

#define GLOBAL_AS __attribute__((address_space(1)))
#define LDS_AS    __attribute__((address_space(3)))

// ---------------- fragment helpers ----------------

static __device__ inline v8h cvt8(float4 a, float4 b) {
  v8h r;
  r[0] = (_Float16)a.x; r[1] = (_Float16)a.y; r[2] = (_Float16)a.z; r[3] = (_Float16)a.w;
  r[4] = (_Float16)b.x; r[5] = (_Float16)b.y; r[6] = (_Float16)b.z; r[7] = (_Float16)b.w;
  return r;
}

static __device__ inline v16h cat16(v8h lo, v8h hi) {
  return __builtin_shufflevector(lo, hi, 0,1,2,3,4,5,6,7,8,9,10,11,12,13,14,15);
}

// A-fragment from an f32 row: 8 halves at row+o0, 8 halves at row+o1
// (contiguous K as called: lane<16 -> K {0..15}, lane>=16 -> K {16..31})
static __device__ inline v16h afrag_f32(const float* __restrict__ row, int o0, int o1) {
  const float4* p0 = (const float4*)(row + o0);
  const float4* p1 = (const float4*)(row + o1);
  return cat16(cvt8(p0[0], p0[1]), cvt8(p1[0], p1[1]));
}

// A-fragment from an f16 row
static __device__ inline v16h afrag_f16(const _Float16* __restrict__ row, int o0, int o1) {
  v8h lo = *(const v8h*)(row + o0);
  v8h hi = *(const v8h*)(row + o1);
  return cat16(lo, hi);
}

static __device__ inline v8f wmma16(v16h a, v16h b, v8f c) {
  return __builtin_amdgcn_wmma_f32_16x16x32_f16(false, a, false, b, (short)0, c,
                                                false, false);
}

static __device__ inline v8f zero8() {
  v8f z;
#pragma unroll
  for (int i = 0; i < 8; ++i) z[i] = 0.0f;
  return z;
}

// ---------------- kernel 0: one-time f32 -> f16 weight conversion ----------
__global__ __launch_bounds__(256)
void cvt_f16_kernel(const float* __restrict__ in, _Float16* __restrict__ out,
                    int n) {
  const int i = (blockIdx.x * 256 + threadIdx.x) * 8;
  if (i + 8 <= n) {
    const float4* p = (const float4*)(in + i);
    *(v8h*)(out + i) = cvt8(p[0], p[1]);
  }
}

// ---------------- kernel 1: QKV projection (fp32 X, f16 W, f16 out) --------
// Y = X @ W^T + b ; MODE 0: Q (scaled by 1/8, [b,h,s,dh])
//                   MODE 1: K ([b,h,s,dh])
//                   MODE 2: V (transposed [b,h,dh,s] so P@V B-frags are contiguous)
// Wave tile: 32(m) x 64(n), K-chunks of 32 -> 8 WMMAs per chunk, B-frag reused 2x.
template <int MODE>
__global__ __launch_bounds__(256)
void mha_proj_kernel(const float* __restrict__ x, const _Float16* __restrict__ w,
                     const float* __restrict__ bias, _Float16* __restrict__ out) {
  const int lane = threadIdx.x & 31;
  const int wave = threadIdx.x >> 5;
  const int wlin = blockIdx.x * 8 + wave;        // 0..2047
  const int n0   = (wlin & 15) * 64;             // 16 n-tiles of 64
  const int m0   = (wlin >> 4) * 32;             // 128 m-tiles of 32
  const int half = lane >> 4;
  const int l16  = lane & 15;

  const float* arow0 = x + (size_t)(m0 + l16) * DEMB;
  const float* arow1 = x + (size_t)(m0 + 16 + l16) * DEMB;

  v8f acc[2][4];
#pragma unroll
  for (int g = 0; g < 2; ++g)
#pragma unroll
    for (int c = 0; c < 4; ++c) acc[g][c] = zero8();

  for (int k0 = 0; k0 < DEMB; k0 += 32) {
    // branch-free, always in-bounds near-cache prefetch of upcoming A data
    const int kpf = (k0 + 128 < DEMB) ? (k0 + 128) : (DEMB - 32);
    __builtin_prefetch((const void*)(arow0 + kpf), 0, 3);
    __builtin_prefetch((const void*)(arow1 + kpf), 0, 3);

    v16h a0 = afrag_f32(arow0, k0 + half * 16, k0 + half * 16 + 8);
    v16h a1 = afrag_f32(arow1, k0 + half * 16, k0 + half * 16 + 8);
#pragma unroll
    for (int c = 0; c < 4; ++c) {
      const _Float16* brow =
          w + (size_t)(n0 + c * 16 + l16) * DEMB + k0 + half * 16;
      const v16h b = *(const v16h*)brow;
      acc[0][c] = wmma16(a0, b, acc[0][c]);
      acc[1][c] = wmma16(a1, b, acc[1][c]);
    }
  }

#pragma unroll
  for (int c = 0; c < 4; ++c) {
    const int n  = n0 + c * 16 + l16;
    const int h  = n >> 6;
    const int dh = n & 63;
    const float bv = bias[n];
#pragma unroll
    for (int g = 0; g < 2; ++g) {
#pragma unroll
      for (int r = 0; r < 8; ++r) {
        const int m = m0 + g * 16 + r + half * 8;  // C/D layout row mapping
        float val = acc[g][c][r] + bv;
        if (MODE == 0) val *= SCALE_INV;           // compile-time
        const int b = m >> 11, s = m & 2047;
        const size_t bh = (size_t)b * NHEAD + h;
        if (MODE == 2)                              // compile-time
          out[(bh * DHEAD + dh) * SEQ + s] = (_Float16)val;
        else
          out[(bh * SEQ + s) * DHEAD + dh] = (_Float16)val;
      }
    }
  }
}

// ---------------- kernel 2: causal flash attention ----------------
// One wave per (b,h, 16-query tile). 32 keys per step: 4 score WMMAs,
// online softmax (shfl_xor width-16 row reductions), P staged via LDS,
// 4 context WMMAs against V^T. Scale already folded into Q.
// K-tiles (contiguous 4KB in Kh) are double-buffered into LDS with
// GLOBAL_LOAD_ASYNC_TO_LDS_B128; s_wait_asynccnt gates consumption.
__global__ __launch_bounds__(256)
void mha_attn_kernel(const _Float16* __restrict__ Qh,
                     const _Float16* __restrict__ Kh,
                     const _Float16* __restrict__ Vt,
                     _Float16* __restrict__ ctx) {
  __shared__ _Float16 plds[8][16 * 32];            // per-wave P tile, 8KB/block
#if HAVE_ASYNC_LDS
  __shared__ _Float16 kstage[8][2][32 * DHEAD];    // per-wave 2x4KB K stage, 64KB
#endif

  const int lane = threadIdx.x & 31;
  const int wave = threadIdx.x >> 5;
  const int wlin = blockIdx.x * 8 + wave;        // 0..4095
  const int q0   = (wlin & 127) * 16;            // 128 query tiles
  const int bh   = wlin >> 7;                    // 0..31 (b*16+h)
  const int half = lane >> 4;
  const int l16  = lane & 15;

  const _Float16* Qbase = Qh + (size_t)bh * SEQ * DHEAD;
  const _Float16* Kbase = Kh + (size_t)bh * SEQ * DHEAD;
  const _Float16* Vbase = Vt + (size_t)bh * DHEAD * SEQ;

  // Q A-fragments for dh chunks [0,32) and [32,64) -- loaded once
  const _Float16* qrow = Qbase + (size_t)(q0 + l16) * DHEAD;
  const v16h aq0 = afrag_f16(qrow, 0 + half * 16, 8 + half * 16);
  const v16h aq1 = afrag_f16(qrow, 32 + half * 16, 40 + half * 16);

  float m_i[8], l_i[8];
  v8f o[4];
#pragma unroll
  for (int r = 0; r < 8; ++r) { m_i[r] = -1e30f; l_i[r] = 0.0f; }
#pragma unroll
  for (int c = 0; c < 4; ++c) o[c] = zero8();

  _Float16* pl = plds[wave];
  const int nsteps = (q0 + 16 + 31) / 32;        // causal: keys [0, q0+16)

#if HAVE_ASYNC_LDS
  // async copy of one contiguous 4KB K-tile: 8 x (32 lanes x 16B)
  auto issue_ktile = [&](int key0, int buf) {
    const char* gsrc = (const char*)(Kbase + (size_t)key0 * DHEAD);
    char* ldst = (char*)&kstage[wave][buf][0];
#pragma unroll
    for (int i = 0; i < 8; ++i) {
      const int off = i * 512 + lane * 16;
      __builtin_amdgcn_global_load_async_to_lds_b128(
          (GLOBAL_AS v4i*)(gsrc + off),
          (LDS_AS v4i*)(ldst + off), 0, 0);
    }
  };
  issue_ktile(0, 0);
#endif

  for (int t = 0; t < nsteps; ++t) {
    const int key0 = t * 32;

#if HAVE_ASYNC_LDS
    if (t + 1 < nsteps) {
      issue_ktile(key0 + 32, (t + 1) & 1);
      // in-order async completion: <=8 outstanding => current tile resident
      asm volatile("s_wait_asynccnt 0x8" ::: "memory");
    } else {
      asm volatile("s_wait_asynccnt 0x0" ::: "memory");
    }
    const _Float16* kb    = &kstage[wave][t & 1][0];
    const _Float16* krow0 = kb + (size_t)l16 * DHEAD;
    const _Float16* krow1 = kb + (size_t)(16 + l16) * DHEAD;
#else
    const _Float16* krow0 = Kbase + (size_t)(key0 + l16) * DHEAD;
    const _Float16* krow1 = Kbase + (size_t)(key0 + 16 + l16) * DHEAD;
#endif

    // ---- scores: S[16q x 32k] = Q(16x64) . K^T ----
    v8f s0 = zero8(), s1 = zero8();
    s0 = wmma16(aq0, *(const v16h*)(krow0 + half * 16), s0);
    s0 = wmma16(aq1, *(const v16h*)(krow0 + 32 + half * 16), s0);
    s1 = wmma16(aq0, *(const v16h*)(krow1 + half * 16), s1);
    s1 = wmma16(aq1, *(const v16h*)(krow1 + 32 + half * 16), s1);

    // ---- causal mask (analytic; only diagonal-adjacent steps need it) ----
    if (key0 + 31 > q0) {
#pragma unroll
      for (int r = 0; r < 8; ++r) {
        const int mq = q0 + r + half * 8;
        if (key0 + l16 > mq)      s0[r] = -1e9f;
        if (key0 + 16 + l16 > mq) s1[r] = -1e9f;
      }
    }

    // ---- online softmax (rows live across 16 lanes of each half-wave) ----
#pragma unroll
    for (int r = 0; r < 8; ++r) {
      float mx = fmaxf(s0[r], s1[r]);
      mx = fmaxf(mx, __shfl_xor(mx, 1, 16));
      mx = fmaxf(mx, __shfl_xor(mx, 2, 16));
      mx = fmaxf(mx, __shfl_xor(mx, 4, 16));
      mx = fmaxf(mx, __shfl_xor(mx, 8, 16));
      const float mnew  = fmaxf(m_i[r], mx);
      const float alpha = __expf(m_i[r] - mnew);
      m_i[r] = mnew;
      const float p0 = __expf(s0[r] - mnew);
      const float p1 = __expf(s1[r] - mnew);
      float rs = p0 + p1;
      rs += __shfl_xor(rs, 1, 16);
      rs += __shfl_xor(rs, 2, 16);
      rs += __shfl_xor(rs, 4, 16);
      rs += __shfl_xor(rs, 8, 16);
      l_i[r] = l_i[r] * alpha + rs;
      o[0][r] *= alpha; o[1][r] *= alpha; o[2][r] *= alpha; o[3][r] *= alpha;
      const int row = r + half * 8;
      pl[row * 32 + l16]      = (_Float16)p0;   // C-layout -> LDS (row-major 16x32)
      pl[row * 32 + 16 + l16] = (_Float16)p1;
    }

    // ---- re-read P as an A-fragment (intra-wave ds RAW; hw waits on DScnt) ----
    const _Float16* prow = pl + l16 * 32;
    const v16h ap = afrag_f16(prow, half * 16, half * 16 + 8);

    // ---- context: O(16x64) += P(16x32) . V(32x64), V stored transposed ----
#pragma unroll
    for (int c = 0; c < 4; ++c) {
      const _Float16* vrow =
          Vbase + (size_t)(c * 16 + l16) * SEQ + key0 + half * 16;
      o[c] = wmma16(ap, *(const v16h*)vrow, o[c]);
    }
  }

  // ---- normalize and store ctx as f16 [b, s, h*64+dh] ----
  const int b = bh >> 4, h = bh & 15;
  _Float16* cb = ctx + (size_t)b * SEQ * DEMB + (size_t)h * DHEAD;
#pragma unroll
  for (int r = 0; r < 8; ++r) {
    const float inv = 1.0f / l_i[r];
    const int s = q0 + r + half * 8;
#pragma unroll
    for (int c = 0; c < 4; ++c)
      cb[(size_t)s * DEMB + c * 16 + l16] = (_Float16)(o[c][r] * inv);
  }
}

// ---------------- kernel 3: output projection (f16 ctx, f16 Wo -> fp32) ----
__global__ __launch_bounds__(256)
void mha_oproj_kernel(const _Float16* __restrict__ ctx,
                      const _Float16* __restrict__ wo,
                      const float* __restrict__ bo, float* __restrict__ out) {
  const int lane = threadIdx.x & 31;
  const int wave = threadIdx.x >> 5;
  const int wlin = blockIdx.x * 8 + wave;        // 0..2047
  const int n0   = (wlin & 15) * 64;
  const int m0   = (wlin >> 4) * 32;
  const int half = lane >> 4;
  const int l16  = lane & 15;

  const _Float16* arow0 = ctx + (size_t)(m0 + l16) * DEMB;
  const _Float16* arow1 = ctx + (size_t)(m0 + 16 + l16) * DEMB;

  v8f acc[2][4];
#pragma unroll
  for (int g = 0; g < 2; ++g)
#pragma unroll
    for (int c = 0; c < 4; ++c) acc[g][c] = zero8();

  for (int k0 = 0; k0 < DEMB; k0 += 32) {
    const int kpf = (k0 + 256 < DEMB) ? (k0 + 256) : (DEMB - 32);
    __builtin_prefetch((const void*)(arow0 + kpf), 0, 3);
    __builtin_prefetch((const void*)(arow1 + kpf), 0, 3);

    v16h a0 = afrag_f16(arow0, k0 + half * 16, k0 + half * 16 + 8);
    v16h a1 = afrag_f16(arow1, k0 + half * 16, k0 + half * 16 + 8);
#pragma unroll
    for (int c = 0; c < 4; ++c) {
      const _Float16* brow =
          wo + (size_t)(n0 + c * 16 + l16) * DEMB + k0 + half * 16;
      const v16h b = *(const v16h*)brow;
      acc[0][c] = wmma16(a0, b, acc[0][c]);
      acc[1][c] = wmma16(a1, b, acc[1][c]);
    }
  }

#pragma unroll
  for (int c = 0; c < 4; ++c) {
    const int n = n0 + c * 16 + l16;
    const float bv = bo[n];
#pragma unroll
    for (int g = 0; g < 2; ++g) {
#pragma unroll
      for (int r = 0; r < 8; ++r) {
        const int m = m0 + g * 16 + r + half * 8;
        out[(size_t)m * DEMB + n] = acc[g][c][r] + bv;
      }
    }
  }
}

// ---------------- launch ----------------
extern "C" void kernel_launch(void* const* d_in, const int* in_sizes, int n_in,
                              void* d_out, int out_size, void* d_ws,
                              size_t ws_size, hipStream_t stream) {
  (void)in_sizes; (void)n_in; (void)out_size; (void)ws_size;

  const float* q   = (const float*)d_in[0];
  const float* k   = (const float*)d_in[1];
  const float* v   = (const float*)d_in[2];
  // d_in[3] = causal mask; applied analytically in the attention kernel.
  const float* w_q = (const float*)d_in[4];
  const float* b_q = (const float*)d_in[5];
  const float* w_k = (const float*)d_in[6];
  const float* b_k = (const float*)d_in[7];
  const float* w_v = (const float*)d_in[8];
  const float* b_v = (const float*)d_in[9];
  const float* w_o = (const float*)d_in[10];
  const float* b_o = (const float*)d_in[11];
  float* out = (float*)d_out;

  const size_t HEAD_ELEMS = (size_t)BATCH * NHEAD * SEQ * DHEAD;  // 4 Mi halves
  const size_t W_ELEMS    = (size_t)DEMB * DEMB;                  // 1 Mi halves
  _Float16* Qh   = (_Float16*)d_ws;
  _Float16* Kh   = Qh + HEAD_ELEMS;
  _Float16* Vt   = Kh + HEAD_ELEMS;
  _Float16* ctx  = Vt + HEAD_ELEMS;
  _Float16* wq16 = ctx + HEAD_ELEMS;
  _Float16* wk16 = wq16 + W_ELEMS;
  _Float16* wv16 = wk16 + W_ELEMS;
  _Float16* wo16 = wv16 + W_ELEMS;   // total 40 MB of workspace

  dim3 blk(256);

  // one-time weight conversion: 1Mi elems, 8 per thread -> 512 blocks each
  cvt_f16_kernel<<<dim3(512), blk, 0, stream>>>(w_q, wq16, (int)W_ELEMS);
  cvt_f16_kernel<<<dim3(512), blk, 0, stream>>>(w_k, wk16, (int)W_ELEMS);
  cvt_f16_kernel<<<dim3(512), blk, 0, stream>>>(w_v, wv16, (int)W_ELEMS);
  cvt_f16_kernel<<<dim3(512), blk, 0, stream>>>(w_o, wo16, (int)W_ELEMS);

  // projections: 2048 waves (128 m-tiles x 16 n-tiles), 8 waves/block
  mha_proj_kernel<0><<<dim3(256), blk, 0, stream>>>(q, wq16, b_q, Qh);
  mha_proj_kernel<1><<<dim3(256), blk, 0, stream>>>(k, wk16, b_k, Kh);
  mha_proj_kernel<2><<<dim3(256), blk, 0, stream>>>(v, wv16, b_v, Vt);

  // attention: 4096 waves (32 bh x 128 q-tiles)
  mha_attn_kernel<<<dim3(512), blk, 0, stream>>>(Qh, Kh, Vt, ctx);

  // output projection: 2048 waves
  mha_oproj_kernel<<<dim3(256), blk, 0, stream>>>(ctx, wo16, b_o, out);
}
